// GCNLSTMLayer_43593918054655
// MI455X (gfx1250) — compile-verified, matrix operand order checked by reading, not run on hardware
//
#include <hip/hip_runtime.h>
#include <math.h>

typedef __attribute__((ext_vector_type(16))) __bf16 v16bf;
typedef __attribute__((ext_vector_type(8)))  float  v8f;

#define ND   64      // feature dim
#define NDEG 32      // neighbors per node
// workspace byte offsets
#define OFF_W0   0        // 256x192 bf16 = 98304 B
#define OFF_W1   98304    // 256x128 bf16 = 65536 B
#define OFF_W2   163840   // 256x128 bf16 = 65536 B
#define OFF_BIAS 229376   // 3x256 f32    = 3072 B
#define OFF_SQ   232448   // N f32

__device__ __forceinline__ __bf16 f2bf(float f) {
    union { float f; unsigned u; } x; x.f = f;
    unsigned r = x.u + 0x7FFFu + ((x.u >> 16) & 1u);   // round-to-nearest-even
    unsigned short hv = (unsigned short)(r >> 16);
    __bf16 o; __builtin_memcpy(&o, &hv, 2);
    return o;
}

__device__ __forceinline__ float fsigmoid(float x) { return 1.0f / (1.0f + __expf(-x)); }
__device__ __forceinline__ float ftanh(float x) {
    float e2 = __expf(2.0f * x);
    return (e2 - 1.0f) / (e2 + 1.0f);
}

// ---- pre-convert weights to bf16 in WMMA-friendly [outCol][K] layout; fuse biases ----
__global__ void prep_kernel(const float* w_ih0, const float* w_hh0, const float* b_ih0, const float* b_hh0,
                            const float* w_ih1, const float* w_hh1, const float* b_ih1, const float* b_hh1,
                            const float* w_ih2, const float* w_hh2, const float* b_ih2, const float* b_hh2,
                            __bf16* Wt0, __bf16* Wt1, __bf16* Wt2, float* bias)
{
    int e = blockIdx.x * 256 + threadIdx.x;
    if (e < 49152) {                                    // layer 0: K = 128(x) + 64(h_prev)
        int n = e / 192, k = e % 192;
        float v = (k < 128) ? w_ih0[n * 128 + k] : w_hh0[n * 64 + (k - 128)];
        Wt0[e] = f2bf(v);
    } else if (e < 49152 + 32768) {                     // layer 1: K = 64 + 64
        int e2 = e - 49152; int n = e2 / 128, k = e2 % 128;
        float v = (k < 64) ? w_ih1[n * 64 + k] : w_hh1[n * 64 + (k - 64)];
        Wt1[e2] = f2bf(v);
    } else if (e < 49152 + 65536) {                     // layer 2
        int e2 = e - 49152 - 32768; int n = e2 / 128, k = e2 % 128;
        float v = (k < 64) ? w_ih2[n * 64 + k] : w_hh2[n * 64 + (k - 64)];
        Wt2[e2] = f2bf(v);
    }
    if (e < 768) {
        int l = e >> 8, j = e & 255;
        const float* bi = (l == 0) ? b_ih0 : (l == 1) ? b_ih1 : b_ih2;
        const float* bh = (l == 0) ? b_hh0 : (l == 1) ? b_hh1 : b_hh2;
        bias[e] = bi[j] + bh[j];
    }
}

// ---- per-source sum of squares (norms monotone in this) ----
__global__ void sumsq_kernel(const float* __restrict__ feature, float* __restrict__ sq, int N)
{
    int n = blockIdx.x * 256 + threadIdx.x;
    if (n >= N) return;
    const float4* f = (const float4*)(feature + (size_t)n * ND);
    float s = 0.0f;
#pragma unroll
    for (int i = 0; i < ND / 4; ++i) {
        float4 v = f[i];
        s += v.x * v.x + v.y * v.y + v.z * v.z + v.w * v.w;
    }
    sq[n] = s;
}

// ---- fused: argmax-gather + 3-layer LSTM step, 32 nodes per workgroup ----
__global__ __launch_bounds__(256, 2)
void fused_kernel(const float* __restrict__ feature, const int* __restrict__ src,
                  const float* __restrict__ h0, const float* __restrict__ c0,
                  const __bf16* __restrict__ Wt0, const __bf16* __restrict__ Wt1,
                  const __bf16* __restrict__ Wt2, const float* __restrict__ bias,
                  const float* __restrict__ sq, float* __restrict__ out, int N)
{
    __shared__ __align__(32) __bf16 lX[32][192];   // A-matrix staging (bf16)
    __shared__ float lG[32][256];                  // gate exchange (f32)
    __shared__ int   lSel[32];
    __shared__ float rv[256];
    __shared__ int   rj[256], rs[256];

    const int tid  = threadIdx.x;
    const int lane = tid & 31;
    const int wv   = tid >> 5;          // 8 waves
    const int r    = wv & 1;            // row tile (16 nodes)
    const int c    = wv >> 1;           // column group (64 gates)
    const int base = blockIdx.x * 32;
    const long N64 = (long)N * 64;

    // ---- phase 1: argmax over neighbor sumsq (8 threads per node, first-max tiebreak) ----
    {
        int ln = tid >> 3, sub = tid & 7;
        int g  = base + ln;
        float bv = -INFINITY; int bj = NDEG, bs = 0;
        if (g < N) {
#pragma unroll
            for (int q = 0; q < NDEG / 8; ++q) {
                int j = sub + q * 8;                       // ascending: strict > keeps first
                int s = src[(size_t)g * NDEG + j];
                float v = sq[s];
                if (v > bv) { bv = v; bj = j; bs = s; }
            }
        }
        rv[tid] = bv; rj[tid] = bj; rs[tid] = bs;
        __syncthreads();
        if (sub == 0 && g < N) {
            float cv = rv[tid]; int cj = rj[tid], cs = rs[tid];
#pragma unroll
            for (int u = 1; u < 8; ++u) {
                float v = rv[tid + u];
                if (v > cv || (v == cv && rj[tid + u] < cj)) { cv = v; cj = rj[tid + u]; cs = rs[tid + u]; }
            }
            lSel[ln] = cs;
        }
        __syncthreads();
    }

    // ---- phase 2: build layer-0 A-matrix rows: [h_sel(64) | feature(64) | h0[0](64)] ----
    for (int e = tid; e < 32 * 192; e += 256) {
        int ln = e / 192, k = e % 192;
        int g = base + ln;
        float v = 0.0f;
        if (g < N) {
            if (k < 64)       v = feature[(size_t)lSel[ln] * ND + k];
            else if (k < 128) v = feature[(size_t)g * ND + (k - 64)];
            else              v = h0[(size_t)g * ND + (k - 128)];
        }
        lX[ln][k] = f2bf(v);
    }
    __syncthreads();

    // ---- phase 3: three LSTM layers ----
    for (int l = 0; l < 3; ++l) {
        const int KT = (l == 0) ? 192 : 128;
        const __bf16* Wt = (l == 0) ? Wt0 : (l == 1) ? Wt1 : Wt2;
        const float* bl = bias + l * 256;

        // warm weight rows into cache (global_prefetch_b8)
        __builtin_prefetch(Wt + (size_t)tid * KT, 0, 1);

        // --- GEMM: 16x64 gate tile per wave via v_wmma_f32_16x16x32_bf16 ---
        v8f acc[4] = {};
        const int m  = lane & 15;
        const int kh = 16 * (lane >> 4);
        for (int k0 = 0; k0 < KT; k0 += 32) {
            v16bf a = *(const v16bf*)(&lX[r * 16 + m][k0 + kh]);
#pragma unroll
            for (int t = 0; t < 4; ++t) {
                const __bf16* bp = Wt + (size_t)(c * 64 + t * 16 + m) * KT + k0 + kh;
                v16bf b = *(const v16bf*)bp;
                acc[t] = __builtin_amdgcn_wmma_f32_16x16x32_bf16(
                    false, a, false, b, (short)0, acc[t], false, false);
            }
        }
        // --- bias add + scatter C tile (lane 0-15: M=v, lane 16-31: M=v+8) to LDS ---
#pragma unroll
        for (int t = 0; t < 4; ++t) {
            int col = c * 64 + t * 16 + m;
            float bj = bl[col];
#pragma unroll
            for (int v = 0; v < 8; ++v) {
                int row = r * 16 + v + 8 * (lane >> 4);
                lG[row][col] = acc[t][v] + bj;
            }
        }
        __syncthreads();

        // --- elementwise LSTM cell (i,f,g,o = cols 0..63, 64..127, 128..191, 192..255) ---
#pragma unroll
        for (int p = 0; p < 8; ++p) {
            int e  = tid + p * 256;          // 32 nodes * 64 dims
            int ln = e >> 6, d = e & 63;
            int g  = base + ln;
            if (g < N) {
                float ig = lG[ln][d];
                float fg = lG[ln][64 + d];
                float gg = lG[ln][128 + d];
                float og = lG[ln][192 + d];
                float cp = c0[(size_t)l * N64 + (size_t)g * ND + d];
                float cn = fsigmoid(fg) * cp + fsigmoid(ig) * ftanh(gg);
                float hn = fsigmoid(og) * ftanh(cn);
                long off = (size_t)l * N64 + (size_t)g * ND + d;
                out[N64 + off]       = hn;   // hs stack
                out[4 * N64 + off]   = cn;   // cs stack
                if (l == 2) out[(size_t)g * ND + d] = hn;   // seq output == hs[2]
                if (l < 2) {                 // next layer A rows: [h_new | h0[l+1]]
                    lX[ln][d]      = f2bf(hn);
                    lX[ln][64 + d] = f2bf(h0[(size_t)(l + 1) * N64 + (size_t)g * ND + d]);
                }
            }
        }
        __syncthreads();
    }
}

extern "C" void kernel_launch(void* const* d_in, const int* in_sizes, int n_in,
                              void* d_out, int out_size, void* d_ws, size_t ws_size,
                              hipStream_t stream) {
    const float* feature = (const float*)d_in[0];
    const int*   src     = (const int*)d_in[1];
    const float* h0      = (const float*)d_in[2];
    const float* c0      = (const float*)d_in[3];
    const float* w_ih0 = (const float*)d_in[4],  * w_hh0 = (const float*)d_in[5];
    const float* b_ih0 = (const float*)d_in[6],  * b_hh0 = (const float*)d_in[7];
    const float* w_ih1 = (const float*)d_in[8],  * w_hh1 = (const float*)d_in[9];
    const float* b_ih1 = (const float*)d_in[10], * b_hh1 = (const float*)d_in[11];
    const float* w_ih2 = (const float*)d_in[12], * w_hh2 = (const float*)d_in[13];
    const float* b_ih2 = (const float*)d_in[14], * b_hh2 = (const float*)d_in[15];

    const int N = in_sizes[0] / ND;
    char* ws = (char*)d_ws;
    __bf16* Wt0 = (__bf16*)(ws + OFF_W0);
    __bf16* Wt1 = (__bf16*)(ws + OFF_W1);
    __bf16* Wt2 = (__bf16*)(ws + OFF_W2);
    float*  bias = (float*)(ws + OFF_BIAS);
    float*  sq   = (float*)(ws + OFF_SQ);

    prep_kernel<<<448, 256, 0, stream>>>(w_ih0, w_hh0, b_ih0, b_hh0,
                                         w_ih1, w_hh1, b_ih1, b_hh1,
                                         w_ih2, w_hh2, b_ih2, b_hh2,
                                         Wt0, Wt1, Wt2, bias);
    sumsq_kernel<<<(N + 255) / 256, 256, 0, stream>>>(feature, sq, N);
    fused_kernel<<<(N + 31) / 32, 256, 0, stream>>>(feature, src, h0, c0,
                                                    Wt0, Wt1, Wt2, bias, sq,
                                                    (float*)d_out, N);
}
